// NodeModel_43954695308053
// MI455X (gfx1250) — matrix operand order repaired and checked by reading, not compiled, not run on hardware
//
#include <hip/hip_runtime.h>
#include <hip/hip_bf16.h>

// ---------------------------------------------------------------------------
// Problem constants (match reference)
// ---------------------------------------------------------------------------
#define NN       100000
#define EE       1600000
#define NODE_IN  128
#define EDGE_IN  128
#define HID      256
#define OUTD     128
#define IN_DIM   512          // NODE_IN + 3*EDGE_IN
#define LN_EPS   1e-5f
#define ROWS     32           // nodes per MLP block

// ---------------------------------------------------------------------------
// Raw clang ext-vector types (avoid HIP vector classes inside unions)
// ---------------------------------------------------------------------------
typedef __attribute__((ext_vector_type(16))) __bf16 v16bf;
typedef __attribute__((ext_vector_type(8)))  float  v8f;
typedef __attribute__((ext_vector_type(4)))  unsigned int u32x4;
typedef __attribute__((ext_vector_type(4)))  float  f32x4;

union FragBF { v16bf v; u32x4 q[2]; };

__device__ __forceinline__ unsigned short f32_to_bf16_rne(float f) {
    unsigned int u = __float_as_uint(f);
    unsigned int r = u + 0x7FFFu + ((u >> 16) & 1u);
    return (unsigned short)(r >> 16);
}

// Monotonic order-preserving f32 <-> u32 mapping: branch-free atomic float max
// via a single u32 atomicMax per element.
__device__ __forceinline__ unsigned int mapF32(float f) {
    unsigned int u = __float_as_uint(f);
    return (u & 0x80000000u) ? ~u : (u | 0x80000000u);
}
__device__ __forceinline__ float unmapF32(unsigned int m) {
    return __uint_as_float((m & 0x80000000u) ? (m ^ 0x80000000u) : ~m);
}

// ---------------------------------------------------------------------------
// Kernel 1: initialize aggregation scratch (must run every launch)
// ---------------------------------------------------------------------------
__global__ void init_agg(float* __restrict__ sumB,
                         unsigned int* __restrict__ maxB,
                         float* __restrict__ cntB) {
    const int total = NN * EDGE_IN;
    for (int i = blockIdx.x * blockDim.x + threadIdx.x; i < total;
         i += gridDim.x * blockDim.x) {
        sumB[i] = 0.0f;
        maxB[i] = 0u;          // below every mapped real float
        if (i < NN) cntB[i] = 0.0f;
    }
}

// ---------------------------------------------------------------------------
// Kernel 2: convert a row-major f32 weight [K, Nout] into bf16 WMMA-B blocked
// layout.  Blocked index = (((nt*nKt + kt)*32 + lane)*16 + i), holding
// W[kt*32 + (lane/16)*16 + i, nt*16 + lane%16]  (contiguous-k B order, same as the A loads).
// ---------------------------------------------------------------------------
__global__ void prep_weights(const float* __restrict__ W,
                             unsigned short* __restrict__ Wb,
                             int K, int Nout) {
    const int nKt = K >> 5;
    const int total = K * Nout;
    for (int idx = blockIdx.x * blockDim.x + threadIdx.x; idx < total;
         idx += gridDim.x * blockDim.x) {
        int i    = idx & 15;
        int lane = (idx >> 4) & 31;
        int tile = idx >> 9;          // nt*nKt + kt
        int kt   = tile % nKt;
        int nt   = tile / nKt;
        int k    = kt * 32 + (lane >> 4) * 16 + i;
        int n    = nt * 16 + (lane & 15);
        Wb[idx]  = f32_to_bf16_rne(W[(size_t)k * Nout + n]);
    }
}

// ---------------------------------------------------------------------------
// Kernel 3: edge scatter aggregation.  One wave32 per edge: 32 lanes x float4
// covers all 128 edge features.  sum via f32 atomics, max via mapped u32 max.
// ---------------------------------------------------------------------------
__global__ __launch_bounds__(256)
void edge_agg(const int* __restrict__ edge_index,
              const float* __restrict__ edge_attr,
              float* __restrict__ sumB, unsigned int* __restrict__ maxB,
              float* __restrict__ cntB) {
    const int lane   = threadIdx.x & 31;
    const int wave   = (blockIdx.x * blockDim.x + threadIdx.x) >> 5;
    const int nWaves = (gridDim.x * blockDim.x) >> 5;
    for (int e = wave; e < EE; e += nWaves) {
        const int col = edge_index[EE + e];          // edge_index[1][e]
        f32x4 v = *(const f32x4*)(edge_attr + (size_t)e * EDGE_IN + lane * 4);
        float*        sp = sumB + (size_t)col * EDGE_IN + lane * 4;
        unsigned int* mp = maxB + (size_t)col * EDGE_IN + lane * 4;
#pragma unroll
        for (int j = 0; j < 4; ++j) {
            atomicAdd(sp + j, v[j]);
            atomicMax(mp + j, mapF32(v[j]));
        }
        if (lane == 0) atomicAdd(cntB + col, 1.0f);
    }
}

// ---------------------------------------------------------------------------
// Fused MLP helpers
// ---------------------------------------------------------------------------
// A fragment (16-bit A, 16x32): lane L holds row m = L%16.
//   elements [0..7]  = K  kt*32 + (L/16)*16 + (0..7)
//   elements [8..15] = K  kt*32 + (L/16)*16 + 8 + (0..7)
// Loop order: A/B fragments each loaded ONCE per k-step and reused across the
// 2 M-tiles x tpw N-tiles of accumulators (halves DS + VMEM traffic per WMMA).
template<int K, int NOUT>
__device__ __forceinline__ void gemm_layer(const unsigned short* __restrict__ stA,
                                           const unsigned short* __restrict__ Wb,
                                           const float* __restrict__ bias,
                                           float* __restrict__ dst, int dstStride,
                                           int lane, int wave) {
    constexpr int nKt = K / 32;
    constexpr int tpw = (NOUT / 16) / 8;   // N-tiles per wave (8 waves)
    const unsigned short* arow0 = stA + (lane & 15) * K;          // M-tile 0
    const unsigned short* arow1 = stA + ((lane & 15) + 16) * K;   // M-tile 1
    const int koff = (lane >> 4) * 16;

    v8f c[2][tpw] = {};
#pragma unroll 4
    for (int kt = 0; kt < nKt; ++kt) {
        FragBF a0, a1;
        a0.q[0] = *(const u32x4*)(arow0 + kt * 32 + koff);
        a0.q[1] = *(const u32x4*)(arow0 + kt * 32 + 8 + koff);
        a1.q[0] = *(const u32x4*)(arow1 + kt * 32 + koff);
        a1.q[1] = *(const u32x4*)(arow1 + kt * 32 + 8 + koff);
#pragma unroll
        for (int ti = 0; ti < tpw; ++ti) {
            const int nt = wave * tpw + ti;
            FragBF b;
            const u32x4* wq =
                (const u32x4*)Wb + ((size_t)nt * nKt + kt) * 64 + lane * 2;
            b.q[0] = wq[0];
            b.q[1] = wq[1];
            c[0][ti] = __builtin_amdgcn_wmma_f32_16x16x32_bf16(
                           false, a0.v, false, b.v, (short)0, c[0][ti], false, false);
            c[1][ti] = __builtin_amdgcn_wmma_f32_16x16x32_bf16(
                           false, a1.v, false, b.v, (short)0, c[1][ti], false, false);
        }
    }
    const int mBase = (lane >> 4) * 8;
#pragma unroll
    for (int ti = 0; ti < tpw; ++ti) {
        const int   ncol = (wave * tpw + ti) * 16 + (lane & 15);
        const float bv   = bias[ncol];
#pragma unroll
        for (int mi = 0; mi < 2; ++mi)
#pragma unroll
            for (int r = 0; r < 8; ++r)
                dst[(size_t)(mi * 16 + mBase + r) * dstStride + ncol] =
                    c[mi][ti][r] + bv;
    }
}

// LayerNorm + SiLU over a [ROWS x 256] f32 LDS tile -> bf16 LDS tile.
// 8 waves, ROWS/8 rows each; 8 features per lane; wave32 shuffle reductions.
__device__ __forceinline__ void ln_silu(const float* __restrict__ stC,
                                        unsigned short* __restrict__ stA,
                                        const float* __restrict__ g,
                                        const float* __restrict__ be,
                                        int lane, int wave) {
#pragma unroll
    for (int rr = 0; rr < ROWS / 8; ++rr) {
        const int m = wave * (ROWS / 8) + rr;
        float v[8];
        float s = 0.0f;
#pragma unroll
        for (int i = 0; i < 8; ++i) {
            v[i] = stC[m * HID + lane + 32 * i];
            s += v[i];
        }
#pragma unroll
        for (int off = 16; off > 0; off >>= 1) s += __shfl_xor(s, off, 32);
        const float mean = s * (1.0f / HID);
        float q = 0.0f;
#pragma unroll
        for (int i = 0; i < 8; ++i) { float d = v[i] - mean; q += d * d; }
#pragma unroll
        for (int off = 16; off > 0; off >>= 1) q += __shfl_xor(q, off, 32);
        const float rstd = rsqrtf(q * (1.0f / HID) + LN_EPS);
#pragma unroll
        for (int i = 0; i < 8; ++i) {
            const int cix = lane + 32 * i;
            float h = (v[i] - mean) * rstd * g[cix] + be[cix];
            float sig = 1.0f / (1.0f + __expf(-h));
            stA[m * HID + cix] = f32_to_bf16_rne(h * sig);
        }
    }
}

// ---------------------------------------------------------------------------
// Kernel 4: fused 4-layer MLP.  One block = ROWS nodes, 256 threads (8 wave32).
// ---------------------------------------------------------------------------
__global__ __launch_bounds__(256)
void mlp_fused(const float* __restrict__ x,
               const float* __restrict__ sumB,
               const unsigned int* __restrict__ maxB,
               const float* __restrict__ cntB,
               const unsigned short* __restrict__ W0b,
               const unsigned short* __restrict__ W1b,
               const unsigned short* __restrict__ W2b,
               const unsigned short* __restrict__ W3b,
               const float* __restrict__ b0, const float* __restrict__ g0,
               const float* __restrict__ be0,
               const float* __restrict__ b1, const float* __restrict__ g1,
               const float* __restrict__ be1,
               const float* __restrict__ b2, const float* __restrict__ g2,
               const float* __restrict__ be2,
               const float* __restrict__ b3,
               float* __restrict__ out) {
    __shared__ __align__(16) unsigned short stA[ROWS * IN_DIM]; // bf16 acts (32KB)
    __shared__ __align__(16) float          stC[ROWS * HID];    // f32 pre-LN (32KB)

    const int t       = threadIdx.x;
    const int lane    = t & 31;
    const int wave    = t >> 5;
    const int rowBase = blockIdx.x * ROWS;

    // Stage concat(sum, max, mean, x) -> bf16 LDS [ROWS x 512]
    for (int e = t; e < ROWS * IN_DIM; e += 256) {
        const int m = e >> 9;
        const int k = e & (IN_DIM - 1);
        const int node = rowBase + m;
        float v;
        if (k < 128) {
            v = sumB[(size_t)node * 128 + k];
        } else if (k < 256) {
            const float c = cntB[node];
            const float mx = unmapF32(maxB[(size_t)node * 128 + (k - 128)]);
            v = (c > 0.0f) ? mx : 0.0f;
        } else if (k < 384) {
            const float c = cntB[node];
            v = sumB[(size_t)node * 128 + (k - 256)] / fmaxf(c, 1.0f);
        } else {
            v = x[(size_t)node * 128 + (k - 384)];
        }
        stA[e] = f32_to_bf16_rne(v);
    }
    __syncthreads();

    gemm_layer<IN_DIM, HID>(stA, W0b, b0, stC, HID, lane, wave);
    __syncthreads();
    ln_silu(stC, stA, g0, be0, lane, wave);
    __syncthreads();

    gemm_layer<HID, HID>(stA, W1b, b1, stC, HID, lane, wave);
    __syncthreads();
    ln_silu(stC, stA, g1, be1, lane, wave);
    __syncthreads();

    gemm_layer<HID, HID>(stA, W2b, b2, stC, HID, lane, wave);
    __syncthreads();
    ln_silu(stC, stA, g2, be2, lane, wave);
    __syncthreads();

    gemm_layer<HID, OUTD>(stA, W3b, b3, out + (size_t)rowBase * OUTD, OUTD,
                          lane, wave);
}

// ---------------------------------------------------------------------------
// Host launch
// ---------------------------------------------------------------------------
static inline size_t align256(size_t v) { return (v + 255) & ~(size_t)255; }

extern "C" void kernel_launch(void* const* d_in, const int* in_sizes, int n_in,
                              void* d_out, int out_size, void* d_ws, size_t ws_size,
                              hipStream_t stream) {
    const float* x          = (const float*)d_in[0];
    const int*   edge_index = (const int*)  d_in[1];
    const float* edge_attr  = (const float*)d_in[2];
    // d_in[3] = u, d_in[4] = batch : unused by the reference math
    const float* W0 = (const float*)d_in[5];
    const float* b0 = (const float*)d_in[6];
    const float* g0 = (const float*)d_in[7];
    const float* be0= (const float*)d_in[8];
    const float* W1 = (const float*)d_in[9];
    const float* b1 = (const float*)d_in[10];
    const float* g1 = (const float*)d_in[11];
    const float* be1= (const float*)d_in[12];
    const float* W2 = (const float*)d_in[13];
    const float* b2 = (const float*)d_in[14];
    const float* g2 = (const float*)d_in[15];
    const float* be2= (const float*)d_in[16];
    const float* W3 = (const float*)d_in[17];
    const float* b3 = (const float*)d_in[18];
    float* out = (float*)d_out;

    // Workspace carve-up
    char* ws = (char*)d_ws;
    size_t off = 0;
    float*        sumB = (float*)(ws + off);        off = align256(off + (size_t)NN * 128 * 4);
    unsigned int* maxB = (unsigned int*)(ws + off); off = align256(off + (size_t)NN * 128 * 4);
    float*        cntB = (float*)(ws + off);        off = align256(off + (size_t)NN * 4);
    unsigned short* W0b = (unsigned short*)(ws + off); off = align256(off + (size_t)IN_DIM * HID * 2);
    unsigned short* W1b = (unsigned short*)(ws + off); off = align256(off + (size_t)HID * HID * 2);
    unsigned short* W2b = (unsigned short*)(ws + off); off = align256(off + (size_t)HID * HID * 2);
    unsigned short* W3b = (unsigned short*)(ws + off); off = align256(off + (size_t)HID * OUTD * 2);
    (void)ws_size; (void)in_sizes; (void)n_in; (void)out_size;

    // 1) reset aggregation scratch (atomics mutate it every call)
    init_agg<<<8192, 256, 0, stream>>>(sumB, maxB, cntB);

    // 2) weights -> bf16 WMMA-B blocked layout (tiny; resident in L2 after)
    prep_weights<<<512, 256, 0, stream>>>(W0, W0b, IN_DIM, HID);
    prep_weights<<<512, 256, 0, stream>>>(W1, W1b, HID, HID);
    prep_weights<<<512, 256, 0, stream>>>(W2, W2b, HID, HID);
    prep_weights<<<512, 256, 0, stream>>>(W3, W3b, HID, OUTD);

    // 3) edge scatter: one wave per edge
    edge_agg<<<8192, 256, 0, stream>>>(edge_index, edge_attr, sumB, maxB, cntB);

    // 4) fused WMMA MLP: ROWS nodes per block
    mlp_fused<<<NN / ROWS, 256, 0, stream>>>(x, sumB, maxB, cntB,
                                             W0b, W1b, W2b, W3b,
                                             b0, g0, be0, b1, g1, be1,
                                             b2, g2, be2, b3, out);
}
